// VectorQuantizer_57664230916925
// MI455X (gfx1250) — matrix operand order repaired and checked by reading, not compiled, not run on hardware
//
#include <hip/hip_runtime.h>
#include <hip/hip_bf16.h>
#include <float.h>

typedef __attribute__((ext_vector_type(16))) __bf16 v16bf;
typedef __attribute__((ext_vector_type(8)))  __bf16 v8bf;
typedef __attribute__((ext_vector_type(8)))  float  v8f;

#define N_TOK 32768
#define K_CB  8192
#define DIM   512
#define M_BLK 64
#define MT    4              // 16-row m-tiles per block
#define NTILES (K_CB / 16)   // 512
#define LDS_PITCH 520        // 512 + 8 bf16 pad (16B) -> conflict-free strided b128

static __device__ __forceinline__ v16bf wmma_cat(v8bf a, v8bf b) {
    return __builtin_shufflevector(a, b, 0,1,2,3,4,5,6,7,8,9,10,11,12,13,14,15);
}

// ---------------- codebook fp32 -> bf16 hi/lo + ||e||^2 ----------------
__global__ __launch_bounds__(256) void cb_convert_kernel(
    const float* __restrict__ cb, __bf16* __restrict__ hi,
    __bf16* __restrict__ lo, float* __restrict__ e_sq)
{
    __shared__ float red[256];
    const int row = blockIdx.x;
    const int tid = threadIdx.x;
    float s = 0.f;
    for (int k = tid; k < DIM; k += 256) {
        float x = cb[(size_t)row * DIM + k];
        __bf16 h = (__bf16)x;
        __bf16 l = (__bf16)(x - (float)h);
        hi[(size_t)row * DIM + k] = h;
        lo[(size_t)row * DIM + k] = l;
        s += x * x;
    }
    red[tid] = s;
    __syncthreads();
    for (int off = 128; off > 0; off >>= 1) {
        if (tid < off) red[tid] += red[tid + off];
        __syncthreads();
    }
    if (tid == 0) e_sq[row] = red[0];
}

// ---------------- fused GEMM(distance) + argmin + gather + loss ----------------
__global__ __launch_bounds__(256) void vq_main_kernel(
    const float* __restrict__ z_e, const float* __restrict__ codebook,
    const __bf16* __restrict__ cb_hi, const __bf16* __restrict__ cb_lo,
    const float* __restrict__ e_sq,
    float* __restrict__ out_zq, float* __restrict__ out_idx,
    float* __restrict__ accum)
{
    __shared__ __attribute__((aligned(32))) __bf16 sh_hi[M_BLK * LDS_PITCH];
    __shared__ __attribute__((aligned(32))) __bf16 sh_lo[M_BLK * LDS_PITCH];
    __shared__ float cand_val[8][M_BLK];
    __shared__ int   cand_idx[8][M_BLK];
    __shared__ int   best_row[M_BLK];
    __shared__ float red[256];

    const int tid = threadIdx.x;
    const int m0  = blockIdx.x * M_BLK;

    // Stage z_e tile into LDS as bf16 hi/lo (coalesced, uniform over all lanes).
    for (int i = tid; i < M_BLK * DIM; i += 256) {
        int m = i >> 9, k = i & (DIM - 1);
        float x = z_e[(size_t)(m0 + m) * DIM + k];
        __bf16 h = (__bf16)x;
        __bf16 l = (__bf16)(x - (float)h);
        sh_hi[m * LDS_PITCH + k] = h;
        sh_lo[m * LDS_PITCH + k] = l;
    }
    __syncthreads();

    const int wave = tid >> 5;
    const int lane = tid & 31;
    const int g    = lane >> 4;   // half-wave: selects K group
    const int mL   = lane & 15;   // M (for A) / N (for B,C) within tile

    float bval[MT][8];
    int   bidx[MT][8];
#pragma unroll
    for (int t = 0; t < MT; ++t)
#pragma unroll
        for (int r = 0; r < 8; ++r) { bval[t][r] = FLT_MAX; bidx[t][r] = 0; }

    for (int nt = wave; nt < NTILES; nt += 8) {
        const int n = nt * 16 + mL;
        const float esq = e_sq[n];
        const __bf16* bh_ptr = cb_hi + (size_t)n * DIM + 16 * g;
        const __bf16* bl_ptr = cb_lo + (size_t)n * DIM + 16 * g;
        if (nt + 8 < NTILES) { // touch next n-tile into cache (global_prefetch_b8)
            __builtin_prefetch(cb_hi + (size_t)(n + 128) * DIM + 16 * g, 0, 0);
            __builtin_prefetch(cb_lo + (size_t)(n + 128) * DIM + 16 * g, 0, 0);
        }

        v8f acc[MT] = {};
#pragma unroll 2
        for (int k0 = 0; k0 < DIM; k0 += 32) {
            // B element e <-> K = 16*g + e : 32 contiguous bytes per lane
            v16bf bh = *(const v16bf*)(bh_ptr + k0);
            v16bf bl = *(const v16bf*)(bl_ptr + k0);
#pragma unroll
            for (int t = 0; t < MT; ++t) {
                // A element e <-> K = 16*g + e: two 16B LDS chunks
                const __bf16* ah = sh_hi + (t * 16 + mL) * LDS_PITCH + k0 + 16 * g;
                const __bf16* al = sh_lo + (t * 16 + mL) * LDS_PITCH + k0 + 16 * g;
                v16bf Ah = wmma_cat(*(const v8bf*)ah, *(const v8bf*)(ah + 8));
                v16bf Al = wmma_cat(*(const v8bf*)al, *(const v8bf*)(al + 8));
                // z.e ~= hi*hi + hi*lo + lo*hi  (error-compensated bf16 split)
                acc[t] = __builtin_amdgcn_wmma_f32_16x16x32_bf16(
                             false, Ah, false, bh, (short)0, acc[t], false, false);
                acc[t] = __builtin_amdgcn_wmma_f32_16x16x32_bf16(
                             false, Ah, false, bl, (short)0, acc[t], false, false);
                acc[t] = __builtin_amdgcn_wmma_f32_16x16x32_bf16(
                             false, Al, false, bh, (short)0, acc[t], false, false);
            }
        }
        // dist(m, n) = ||e_n||^2 - 2 z_m.e_n   (||z||^2 constant per row)
#pragma unroll
        for (int t = 0; t < MT; ++t)
#pragma unroll
            for (int r = 0; r < 8; ++r) {
                float dist = esq - 2.f * acc[t][r];
                if (dist < bval[t][r]) { bval[t][r] = dist; bidx[t][r] = n; }
            }
    }

    // Butterfly argmin across the 16 lanes sharing a row (xor<=8 stays in half-wave)
#pragma unroll
    for (int t = 0; t < MT; ++t)
#pragma unroll
        for (int r = 0; r < 8; ++r) {
            float v = bval[t][r]; int ix = bidx[t][r];
            for (int off = 1; off < 16; off <<= 1) {
                float v2 = __shfl_xor(v, off, 32);
                int   i2 = __shfl_xor(ix, off, 32);
                if (v2 < v || (v2 == v && i2 < ix)) { v = v2; ix = i2; }
            }
            if (mL == 0) {
                int m = t * 16 + r + 8 * g;   // C layout: VGPR r -> M = r + 8*g
                cand_val[wave][m] = v;
                cand_idx[wave][m] = ix;
            }
        }
    __syncthreads();

    // Reduce 8 wave candidates per row; emit index
    if (tid < M_BLK) {
        float v = cand_val[0][tid]; int ix = cand_idx[0][tid];
#pragma unroll
        for (int w = 1; w < 8; ++w) {
            float v2 = cand_val[w][tid]; int i2 = cand_idx[w][tid];
            if (v2 < v || (v2 == v && i2 < ix)) { v = v2; ix = i2; }
        }
        best_row[tid] = ix;
        out_idx[m0 + tid] = (float)ix;
    }
    __syncthreads();

    // Coalesced gather of z_q + squared-error partial sum
    float s = 0.f;
    for (int i = tid; i < M_BLK * DIM; i += 256) {
        int m = i >> 9, k = i & (DIM - 1);
        int ix = best_row[m];
        float q  = codebook[(size_t)ix * DIM + k];
        float ze = z_e[(size_t)(m0 + m) * DIM + k];
        float d = q - ze;
        s += d * d;
        out_zq[(size_t)(m0 + m) * DIM + k] = q;
    }
    red[tid] = s;
    __syncthreads();
    for (int off = 128; off > 0; off >>= 1) {
        if (tid < off) red[tid] += red[tid + off];
        __syncthreads();
    }
    if (tid == 0) atomicAdd(accum, red[0]);
}

__global__ void finalize_kernel(const float* __restrict__ accum,
                                float* __restrict__ out_losses)
{
    float mean = *accum / (float)((size_t)N_TOK * DIM);
    out_losses[0] = mean;          // codebook loss
    out_losses[1] = 0.25f * mean;  // commitment loss
}

extern "C" void kernel_launch(void* const* d_in, const int* in_sizes, int n_in,
                              void* d_out, int out_size, void* d_ws, size_t ws_size,
                              hipStream_t stream) {
    const float* z_e      = (const float*)d_in[0];
    const float* codebook = (const float*)d_in[1];

    // workspace layout
    __bf16* cb_hi = (__bf16*)d_ws;                       // 8 MB
    __bf16* cb_lo = cb_hi + (size_t)K_CB * DIM;          // 8 MB
    float*  e_sq  = (float*)(cb_lo + (size_t)K_CB * DIM);// 32 KB
    float*  accum = e_sq + K_CB;                         // 4 B

    float* out_zq     = (float*)d_out;                   // N*D
    float* out_idx    = out_zq + (size_t)N_TOK * DIM;    // N
    float* out_losses = out_idx + N_TOK;                 // 2

    hipMemsetAsync(accum, 0, sizeof(float), stream);

    cb_convert_kernel<<<K_CB, 256, 0, stream>>>(codebook, cb_hi, cb_lo, e_sq);
    vq_main_kernel<<<N_TOK / M_BLK, 256, 0, stream>>>(
        z_e, codebook, cb_hi, cb_lo, e_sq, out_zq, out_idx, accum);
    finalize_kernel<<<1, 1, 0, stream>>>(accum, out_losses);
}
